// AttentionBlock_60739427500720
// MI455X (gfx1250) — matrix operand order repaired and checked by reading, not compiled, not run on hardware
//
#include <hip/hip_runtime.h>

// ---------------------------------------------------------------------------
// AttentionBlock for MI455X (gfx1250, wave32, WMMA + TDM).
// B=8, C=32, F=32, T=4000, Ch=16, D=Ch*F=512.
//
//   k1 qkv   : x[B,C,F,T] -> q_tok[B,T,D] bf16, k_tok[B,T,D] bf16 (/sqrt(D)),
//              v_nat[B,D,T] bf16          (conv 1x1 + BN + PReLU fused)
//   k2 attn  : block = 64 query rows, 8 waves. Keys streamed in chunks of 32,
//              staged by the Tensor Data Mover into double-buffered LDS.
//              S = Q*K^T (v_wmma_f32_16x16x32_bf16, 4-way split accumulators),
//              sigmoid gate, O += P*V. O -> sa_tok[B,T,D] f32.
//   k3 enc   : conv 1x1 + BN + PReLU + residual -> out[B,C,F,T] f32.
// ---------------------------------------------------------------------------

#define Bn 8
#define Cn 32
#define Fn 32
#define Tn 4000
#define CH 16
#define Dn 512
#define NCH (Tn / 32)                       // 125 key chunks
#define NBLK ((Tn + 63) / 64)               // 63 query blocks (last is ragged)
#define EPS 1e-5f
#define INV_SQRT_D 0.044194173824159216f    // 1/sqrt(512)

typedef __attribute__((ext_vector_type(16))) __bf16 v16bf;
typedef __attribute__((ext_vector_type(8)))  float  v8f;
typedef __attribute__((ext_vector_type(4)))  unsigned int u32x4;
typedef __attribute__((ext_vector_type(8)))  int i32x8;
typedef __attribute__((ext_vector_type(4)))  int i32x4;

#if __has_builtin(__builtin_amdgcn_tensor_load_to_lds) && \
    __has_builtin(__builtin_amdgcn_s_wait_tensorcnt)
#define USE_TDM 1
#else
#define USE_TDM 0
#endif

struct U32x8s { uint4 lo, hi; };

static __device__ __forceinline__ v16bf pack_ab(uint4 lo, uint4 hi) {
  U32x8s u{lo, hi};
  return __builtin_bit_cast(v16bf, u);
}
static __device__ __forceinline__ uint4 lds_u4(const __bf16* p) {
  return *reinterpret_cast<const uint4*>(p);
}
static __device__ __forceinline__ float fast_sigmoid(float x) {
  float e = __builtin_amdgcn_exp2f(-1.4426950408889634f * x);
  return 1.0f / (1.0f + e);
}

#if USE_TDM
// 2-D TDM load: tile (tile_d1 rows x tile_d0 elems) of a row-major tensor
// (tensor_d1 x tensor_d0, row stride stride0, 2-byte elements) -> LDS,
// packed contiguously.  D# bitfields per CDNA5 ISA 8.3/8.4.
// 6-arg toolchain variant: (g0, g1, g2, g3, extra, cpol); groups 2/3 unused
// for 2-D tiles (tile_dim2 = 0) -> zero-filled.
static __device__ __forceinline__ void tdm_load_2d(
    unsigned lds_addr, unsigned long long gaddr,
    unsigned tensor_d0, unsigned tensor_d1, unsigned long long stride0,
    unsigned tile_d0, unsigned tile_d1)
{
  u32x4 g0;
  g0[0] = 1u;                                            // count=1 valid D#
  g0[1] = lds_addr;                                      // LDS byte address
  g0[2] = (unsigned)(gaddr & 0xFFFFFFFFull);             // global_addr[31:0]
  g0[3] = (unsigned)((gaddr >> 32) & 0x1FFFFFFull)       // global_addr[56:32]
          | (2u << 30);                                  // type = 2 (image)
  i32x8 g1;
  g1[0] = (int)(1u << 16);                               // data_size = 2 bytes
  g1[1] = (int)((tensor_d0 & 0xFFFFu) << 16);            // tensor_dim0[15:0]
  g1[2] = (int)(((tensor_d0 >> 16) & 0xFFFFu) | ((tensor_d1 & 0xFFFFu) << 16));
  g1[3] = (int)(((tensor_d1 >> 16) & 0xFFFFu) | ((tile_d0 & 0xFFFFu) << 16));
  g1[4] = (int)(tile_d1 & 0xFFFFu);                      // tile_dim1, tile_dim2=0
  g1[5] = (int)(unsigned)(stride0 & 0xFFFFFFFFull);      // dim0_stride[31:0]
  g1[6] = (int)(unsigned)((stride0 >> 32) & 0xFFFFull);  // dim0_stride[47:32]
  g1[7] = 0;
  i32x4 z4 = {0, 0, 0, 0};
  i32x8 z8 = {0, 0, 0, 0, 0, 0, 0, 0};
  __builtin_amdgcn_tensor_load_to_lds(g0, g1, z4, z4, z8, 0);
}
#endif

// ---------------------------------------------------------------------------
// Kernel 1: qkv conv blocks (1x1 conv + inference BN + PReLU)
// ---------------------------------------------------------------------------
__global__ __launch_bounds__(256) void qkv_kernel(
    const float* __restrict__ x,
    const float* __restrict__ W, const float* __restrict__ bias,
    const float* __restrict__ gamma, const float* __restrict__ beta,
    const float* __restrict__ mean, const float* __restrict__ var,
    const float* __restrict__ alpha,
    __bf16* __restrict__ q_tok, __bf16* __restrict__ k_tok,
    __bf16* __restrict__ v_nat)
{
  int idx = blockIdx.x * 256 + threadIdx.x;
  int t = idx % Tn;
  int rem = idx / Tn;
  int f = rem & (Fn - 1);
  int b = rem >> 5;

  float xv[Cn];
#pragma unroll
  for (int i = 0; i < Cn; ++i)
    xv[i] = x[((b * Cn + i) * Fn + f) * Tn + t];

#pragma unroll
  for (int n = 0; n < 3; ++n) {
    float a_n = alpha[n];
#pragma unroll
    for (int c = 0; c < CH; ++c) {
      int pc = n * CH + c;
      const float* wr = W + pc * Cn;
      float h = bias[pc];
#pragma unroll
      for (int i = 0; i < Cn; ++i) h = fmaf(wr[i], xv[i], h);
      float inv = gamma[pc] * rsqrtf(var[pc] + EPS);
      float y = (h - mean[pc]) * inv + beta[pc];
      y = y > 0.0f ? y : a_n * y;
      int d = c * Fn + f;
      if (n == 0)      q_tok[(b * Tn + t) * Dn + d] = (__bf16)y;
      else if (n == 1) k_tok[(b * Tn + t) * Dn + d] = (__bf16)(y * INV_SQRT_D);
      else             v_nat[(b * Dn + d) * Tn + t] = (__bf16)y;
    }
  }
}

// ---------------------------------------------------------------------------
// Kernel 2: sigmoid-gated attention, streaming keys, TDM-staged tiles.
// Block: 256 threads (8 waves), 64 query rows.
//   S phase: wave w computes S tile (r=w>>1 in 0..3, c=w&1 in 0..1) of the
//            64x32 S block; K-dim = D = 512 -> 16 WMMAs, 4 partial accums.
//   O phase: wave w owns d-columns [w*64, w*64+64): 4x4 tiles, K-dim = 32.
// WMMA 16-bit operand layouts (ISA 7.12.2):
//   A (16x32): lane<16 -> row M=lane, elems = K 0..7 | 16..23
//              lane>=16 -> row M=lane-16, elems = K 8..15 | 24..31
//   B (32x16): lane<16 -> col N=lane; this code loads K 0..15 contiguous, A likewise
//              lane>=16 -> col N=lane-16, K 16..31 (any k order is valid if A and B match)
// ---------------------------------------------------------------------------
__global__ __launch_bounds__(256) void attn_kernel(
    const __bf16* __restrict__ q_tok,
    const __bf16* __restrict__ k_tok,
    const __bf16* __restrict__ v_nat,
    float* __restrict__ sa_tok)
{
  extern __shared__ __align__(16) char smem[];
  __bf16* lq  = (__bf16*)smem;                 // 64 x 512          (64 KB)
  __bf16* lk0 = lq  + 64 * Dn;                 // 32 x 512  buf 0   (32 KB)
  __bf16* lk1 = lk0 + 32 * Dn;                 // 32 x 512  buf 1
  __bf16* lv0 = lk1 + 32 * Dn;                 // 512 x 32  buf 0   ([d][s])
  __bf16* lv1 = lv0 + Dn * 32;                 // 512 x 32  buf 1
  __bf16* lp  = lv1 + Dn * 32;                 // 64 x 32           (4 KB)

  const int tid  = threadIdx.x;
  const int wave = tid >> 5;
  const int lane = tid & 31;
  const int half = lane >> 4;
  const int ln   = lane & 15;
  const int b    = blockIdx.y;
  const int m0   = blockIdx.x * 64;
  const int sr   = wave >> 1;                  // S tile row (0..3)
  const int sc   = wave & 1;                   // S tile col (0..1)

#if USE_TDM
  if (wave == 0) {
    // Q tile (ragged tail handled by tensor_dim1 OOB zero-fill)
    tdm_load_2d((unsigned)(size_t)lq,
                (unsigned long long)(size_t)(q_tok + (size_t)(b * Tn + m0) * Dn),
                Dn, (unsigned)(Tn - m0), Dn, Dn, 64);
    tdm_load_2d((unsigned)(size_t)lk0,
                (unsigned long long)(size_t)(k_tok + (size_t)(b * Tn) * Dn),
                Dn, Tn, Dn, Dn, 32);
    tdm_load_2d((unsigned)(size_t)lv0,
                (unsigned long long)(size_t)(v_nat + (size_t)b * Dn * Tn),
                Tn, Dn, Tn, 32, Dn);
  }
#else
  {
    const uint4* qg = (const uint4*)q_tok;     // 64 uint4 per token row
    uint4* dq = (uint4*)lq;
#pragma unroll
    for (int i = 0; i < 16; ++i) {
      int id = tid + i * 256;
      int row = id >> 6, part = id & 63;
      int rg = m0 + row; if (rg >= Tn) rg = Tn - 1;
      dq[id] = qg[(size_t)(b * Tn + rg) * 64 + part];
    }
  }
#endif

  v8f acc[4][4];
#pragma unroll
  for (int r = 0; r < 4; ++r)
#pragma unroll
    for (int cc = 0; cc < 4; ++cc)
#pragma unroll
      for (int j = 0; j < 8; ++j) acc[r][cc][j] = 0.0f;

  for (int it = 0; it < NCH; ++it) {
    const int s0 = it * 32;
#if USE_TDM
    if (wave == 0) __builtin_amdgcn_s_wait_tensorcnt(0);
    __syncthreads();                           // chunk data visible, old reads done
    const __bf16* lkc = (it & 1) ? lk1 : lk0;
    const __bf16* lvc = (it & 1) ? lv1 : lv0;
    if (wave == 0 && it + 1 < NCH) {           // prefetch next chunk via DMA
      const int s1 = s0 + 32;
      __bf16* lkn = (it & 1) ? lk0 : lk1;
      __bf16* lvn = (it & 1) ? lv0 : lv1;
      tdm_load_2d((unsigned)(size_t)lkn,
                  (unsigned long long)(size_t)(k_tok + (size_t)(b * Tn + s1) * Dn),
                  Dn, (unsigned)(Tn - s1), Dn, Dn, 32);
      tdm_load_2d((unsigned)(size_t)lvn,
                  (unsigned long long)(size_t)(v_nat + (size_t)b * Dn * Tn + s1),
                  (unsigned)(Tn - s1), Dn, Tn, 32, Dn);
    }
#else
    __syncthreads();
    {
      const uint4* kg = (const uint4*)k_tok;
      uint4* dk = (uint4*)lk0;
#pragma unroll
      for (int i = 0; i < 8; ++i) {
        int id = tid + i * 256;
        int row = id >> 6, part = id & 63;
        dk[id] = kg[(size_t)(b * Tn + s0 + row) * 64 + part];
      }
      uint4* dv = (uint4*)lv0;
#pragma unroll
      for (int i = 0; i < 8; ++i) {
        int id = tid + i * 256;
        int d = id >> 2, part = id & 3;
        dv[id] = *(const uint4*)(v_nat + (size_t)(b * Dn + d) * Tn + s0 + part * 8);
      }
    }
    __syncthreads();
    const __bf16* lkc = lk0;
    const __bf16* lvc = lv0;
#endif

    // ---- S = Q * K^T : one 16x16 tile per wave, 4-way split accumulation ----
    v8f sp[4];
#pragma unroll
    for (int p = 0; p < 4; ++p)
#pragma unroll
      for (int j = 0; j < 8; ++j) sp[p][j] = 0.0f;
#pragma unroll
    for (int kk = 0; kk < 16; ++kk) {
      const __bf16* ab = lq + (sr * 16 + ln) * Dn + kk * 32 + half * 16;
      v16bf A = pack_ab(lds_u4(ab), lds_u4(ab + 8));
      const __bf16* bb = lkc + (sc * 16 + ln) * Dn + kk * 32 + half * 16;
      v16bf Bm = pack_ab(lds_u4(bb), lds_u4(bb + 8));
      sp[kk & 3] = __builtin_amdgcn_wmma_f32_16x16x32_bf16(
          false, A, false, Bm, (short)0, sp[kk & 3], false, false);
    }
    v8f sAcc = (sp[0] + sp[1]) + (sp[2] + sp[3]);

    // ---- sigmoid gate -> bf16 P tile in LDS ----
#pragma unroll
    for (int j = 0; j < 8; ++j) {
      float p = fast_sigmoid(sAcc[j]);
      lp[(sr * 16 + j + half * 8) * 32 + sc * 16 + ln] = (__bf16)p;
    }
    __syncthreads();

    // ---- O += P * V : 4 row-tiles x 4 col-tiles per wave, K-dim = 32 ----
#pragma unroll
    for (int r = 0; r < 4; ++r) {
      const __bf16* pb = lp + (r * 16 + ln) * 32 + half * 16;
      v16bf Ap = pack_ab(lds_u4(pb), lds_u4(pb + 8));
#pragma unroll
      for (int cc = 0; cc < 4; ++cc) {
        const __bf16* vb = lvc + (wave * 64 + cc * 16 + ln) * 32 + half * 16;
        v16bf Bv = pack_ab(lds_u4(vb), lds_u4(vb + 8));
        acc[r][cc] = __builtin_amdgcn_wmma_f32_16x16x32_bf16(
            false, Ap, false, Bv, (short)0, acc[r][cc], false, false);
      }
    }
  }

  // ---- write O as sa_tok[b][t][d] f32 (64B segments per lane half) ----
#pragma unroll
  for (int r = 0; r < 4; ++r)
#pragma unroll
    for (int cc = 0; cc < 4; ++cc)
#pragma unroll
      for (int j = 0; j < 8; ++j) {
        int t = m0 + r * 16 + j + half * 8;
        if (t < Tn) {
          int d = wave * 64 + cc * 16 + ln;
          sa_tok[(size_t)(b * Tn + t) * Dn + d] = acc[r][cc][j];
        }
      }
}

// ---------------------------------------------------------------------------
// Kernel 3: enc conv block + residual
// ---------------------------------------------------------------------------
__global__ __launch_bounds__(256) void enc_kernel(
    const float* __restrict__ x,
    const float* __restrict__ sa_tok,
    const float* __restrict__ W, const float* __restrict__ bias,
    const float* __restrict__ gamma, const float* __restrict__ beta,
    const float* __restrict__ mean, const float* __restrict__ var,
    const float* __restrict__ alpha,
    float* __restrict__ out)
{
  int idx = blockIdx.x * 256 + threadIdx.x;
  int t = idx % Tn;
  int rem = idx / Tn;
  int f = rem & (Fn - 1);
  int b = rem >> 5;

  float a = alpha[0];
  float s[CH];
#pragma unroll
  for (int ch = 0; ch < CH; ++ch)
    s[ch] = sa_tok[(size_t)(b * Tn + t) * Dn + ch * Fn + f];

#pragma unroll
  for (int o = 0; o < Cn; ++o) {
    const float* wr = W + o * CH;
    float h = bias[o];
#pragma unroll
    for (int ch = 0; ch < CH; ++ch) h = fmaf(wr[ch], s[ch], h);
    float inv = gamma[o] * rsqrtf(var[o] + EPS);
    float y = (h - mean[o]) * inv + beta[o];
    y = y > 0.0f ? y : a * y;
    int xi = ((b * Cn + o) * Fn + f) * Tn + t;
    out[xi] = y + x[xi];
  }
}

// ---------------------------------------------------------------------------
extern "C" void kernel_launch(void* const* d_in, const int* in_sizes, int n_in,
                              void* d_out, int out_size, void* d_ws, size_t ws_size,
                              hipStream_t stream) {
  const float* x          = (const float*)d_in[0];
  const float* qkv_W      = (const float*)d_in[1];
  const float* qkv_b      = (const float*)d_in[2];
  const float* qkv_gamma  = (const float*)d_in[3];
  const float* qkv_beta   = (const float*)d_in[4];
  const float* qkv_mean   = (const float*)d_in[5];
  const float* qkv_var    = (const float*)d_in[6];
  const float* qkv_alpha  = (const float*)d_in[7];
  const float* enc_W      = (const float*)d_in[8];
  const float* enc_b      = (const float*)d_in[9];
  const float* enc_gamma  = (const float*)d_in[10];
  const float* enc_beta   = (const float*)d_in[11];
  const float* enc_mean   = (const float*)d_in[12];
  const float* enc_var    = (const float*)d_in[13];
  const float* enc_alpha  = (const float*)d_in[14];
  float* out = (float*)d_out;

  const size_t tok = (size_t)Bn * Tn * Dn;
  __bf16* q_tok = (__bf16*)d_ws;
  __bf16* k_tok = q_tok + tok;
  __bf16* v_nat = k_tok + tok;
  float*  sa    = (float*)(v_nat + tok);

  const int total = Bn * Fn * Tn;
  qkv_kernel<<<total / 256, 256, 0, stream>>>(
      x, qkv_W, qkv_b, qkv_gamma, qkv_beta, qkv_mean, qkv_var, qkv_alpha,
      q_tok, k_tok, v_nat);

  // LDS: Q 64K + 2x K 32K + 2x V 32K + P 4K = 200,704 bytes
  const size_t smem =
      (size_t)(64 * Dn + 2 * 32 * Dn + 2 * Dn * 32 + 64 * 32) * sizeof(__bf16);
  attn_kernel<<<dim3(NBLK, Bn), 256, smem, stream>>>(q_tok, k_tok, v_nat, sa);

  enc_kernel<<<total / 256, 256, 0, stream>>>(
      x, sa, enc_W, enc_b, enc_gamma, enc_beta, enc_mean, enc_var, enc_alpha, out);
}
